// SiameseSegNet_89661737271882
// MI455X (gfx1250) — matrix operand order repaired and checked by reading, not compiled, not run on hardware
//
#include <hip/hip_runtime.h>
#include <hip/hip_bf16.h>

// ---------------------------------------------------------------------------
// Types for CDNA5 WMMA (wave32): A 16x32 bf16, B 32x16 bf16, C/D 16x16 f32
// ---------------------------------------------------------------------------
typedef __attribute__((ext_vector_type(16))) __bf16 v16bf;
typedef __attribute__((ext_vector_type(8)))  __bf16 v8bf;
typedef __attribute__((ext_vector_type(8)))  float  v8f;

typedef unsigned short bfbits;

__device__ __forceinline__ bfbits f2bf(float f) {
  __bf16 h = (__bf16)f;
  return __builtin_bit_cast(bfbits, h);
}
__device__ __forceinline__ float bf2f(bfbits u) {
  __bf16 h = __builtin_bit_cast(__bf16, u);
  return (float)h;
}

#define EPS_BN 1e-5f

// ---------------------------------------------------------------------------
// Weight prep: f32 -> bf16, TAP-MAJOR layout dst[(tap*CoutP + co)*CinP + ci].
// Channel padding on BOTH axes: ci >= cinSrc -> 0 (CinP multiple of 32),
// co >= cout -> 0 (CoutP multiple of 64) so A-fragment loads never leave the
// table and padded rows/cols contribute exact zeros.
// flip==1: source is ConvTranspose layout (Cin,Cout,3,3); flip taps and swap
//          in/out so the same conv kernel implements deconv (k3,s1,p1).
// ---------------------------------------------------------------------------
__global__ void prep_weights(const float* __restrict__ src, bfbits* __restrict__ dst,
                             int cinSrc, int cinP, int cout, int coutP, int flip, int n) {
  int i = blockIdx.x * blockDim.x + threadIdx.x;
  if (i >= n) return;
  int ci  = i % cinP;
  int co  = (i / cinP) % coutP;
  int tap = i / (cinP * coutP);
  float v = 0.0f;
  if (ci < cinSrc && co < cout) {
    if (flip) {
      v = src[((size_t)ci * cout + co) * 9 + (8 - tap)];  // (2-dy)*3+(2-dx)=8-tap
    } else {
      v = src[((size_t)co * cinSrc + ci) * 9 + tap];
    }
  }
  dst[i] = f2bf(v);
}

// NCHW f32 image (3,H,W) -> NHWC bf16 padded to 32 channels
__global__ void img_to_nhwc(const float* __restrict__ in, bfbits* __restrict__ out,
                            int HW) {
  int i = blockIdx.x * blockDim.x + threadIdx.x;
  int n = HW * 32;
  if (i >= n) return;
  int c   = i & 31;
  int pix = i >> 5;
  out[i] = (c < 3) ? f2bf(in[(size_t)c * HW + pix]) : (bfbits)0;
}

__global__ void zero_bf(bfbits* __restrict__ p, int n) {
  int i = blockIdx.x * blockDim.x + threadIdx.x;
  if (i < n) p[i] = 0;
}

// ---------------------------------------------------------------------------
// LDS-free implicit-GEMM 3x3 conv (pad 1, stride 1), NHWC activations,
// via v_wmma_f32_16x16x32_bf16.
// Workgroup (256 thr = 8 waves) tile: 64 output channels x 128 pixels.
// Wave tile: 32 co x 32 pixels = 2 A frags x 2 B frags -> 4 WMMA / 8 loads.
// Out-of-image taps read a zero page via branch-free pointer select (no exec
// masking, no register zero-fills in the K loop).  Cin multiple of 32 and
// CoutP multiple of 64 make every vector load 16B/32B aligned and in-bounds.
// Fused bias+BN+ReLU, packed v8bf NHWC stores.
// ---------------------------------------------------------------------------
__global__ __launch_bounds__(256)
void conv3x3_wmma_nhwc(const bfbits* __restrict__ in, const bfbits* __restrict__ wt,
                       const bfbits* __restrict__ zbuf,
                       const float* __restrict__ bias, const float* __restrict__ gamma,
                       const float* __restrict__ beta, bfbits* __restrict__ out,
                       int Cin, int Cout, int CoutP, int H, int W,
                       int do_bn, int do_relu) {
  const int tid  = threadIdx.x;
  const int lane = tid & 31;
  const int wave = tid >> 5;
  const int wrow = wave >> 2;   // 0..1 : which 32 output channels
  const int wcol = wave & 3;    // 0..3 : which 32 pixels
  const int g    = lane >> 4;   // half-wave group (ISA layout)
  const int l16  = lane & 15;

  const int pix0 = blockIdx.x * 128;
  const int co0  = blockIdx.y * 64;

  // fixed per-lane coordinates
  const int arow0 = co0 + wrow * 32 + l16;          // A rows (output channels)
  const int arow1 = arow0 + 16;
  const int p0    = pix0 + wcol * 32 + l16;         // pixel for acc*0 columns
  const int p1    = p0 + 16;                        // pixel for acc*1 columns
  const int y0 = p0 / W, x0 = p0 % W;
  const int y1 = p1 / W, x1 = p1 % W;

  const size_t aBase0   = (size_t)arow0 * Cin;      // within one tap plane
  const size_t aBase1   = (size_t)arow1 * Cin;
  const size_t tapPlane = (size_t)CoutP * Cin;

  v8f acc00 = {}, acc01 = {}, acc10 = {}, acc11 = {};

  for (int tap = 0; tap < 9; ++tap) {
    const int dy = tap / 3 - 1;
    const int dx = tap % 3 - 1;

    // per-tap, lane-constant: branch-free zero-page select for border taps
    const int yy0 = y0 + dy, xx0 = x0 + dx;
    const int yy1 = y1 + dy, xx1 = x1 + dx;
    const bool v0 = (unsigned)yy0 < (unsigned)H && (unsigned)xx0 < (unsigned)W;
    const bool v1 = (unsigned)yy1 < (unsigned)H && (unsigned)xx1 < (unsigned)W;
    const bfbits* pb0 = v0 ? in + ((size_t)(yy0 * W + xx0)) * Cin : zbuf;
    const bfbits* pb1 = v1 ? in + ((size_t)(yy1 * W + xx1)) * Cin : zbuf;
    const bfbits* pa0 = wt + (size_t)tap * tapPlane + aBase0;
    const bfbits* pa1 = wt + (size_t)tap * tapPlane + aBase1;

    for (int kb = 0; kb < Cin; kb += 32) {
      // ---- A fragments: K = kb + {g*16..g*16+15} ----
      v8bf a0lo = *(const v8bf*)(pa0 + kb + g * 16);
      v8bf a0hi = *(const v8bf*)(pa0 + kb + g * 16 + 8);
      v8bf a1lo = *(const v8bf*)(pa1 + kb + g * 16);
      v8bf a1hi = *(const v8bf*)(pa1 + kb + g * 16 + 8);
      v16bf a0 = __builtin_shufflevector(a0lo, a0hi, 0, 1, 2, 3, 4, 5, 6, 7,
                                         8, 9, 10, 11, 12, 13, 14, 15);
      v16bf a1 = __builtin_shufflevector(a1lo, a1hi, 0, 1, 2, 3, 4, 5, 6, 7,
                                         8, 9, 10, 11, 12, 13, 14, 15);
      // ---- B fragments: 16 contiguous channels at this lane's pixel ----
      v16bf b0 = *(const v16bf*)(pb0 + kb + g * 16);   // 32B aligned
      v16bf b1 = *(const v16bf*)(pb1 + kb + g * 16);

      acc00 = __builtin_amdgcn_wmma_f32_16x16x32_bf16(false, a0, false, b0,
                                                      (short)0, acc00, false, false);
      acc01 = __builtin_amdgcn_wmma_f32_16x16x32_bf16(false, a0, false, b1,
                                                      (short)0, acc01, false, false);
      acc10 = __builtin_amdgcn_wmma_f32_16x16x32_bf16(false, a1, false, b0,
                                                      (short)0, acc10, false, false);
      acc11 = __builtin_amdgcn_wmma_f32_16x16x32_bf16(false, a1, false, b1,
                                                      (short)0, acc11, false, false);
    }
  }

  // ---- epilogue: bias + BN(eval, mean=0,var=1) + ReLU, NHWC bf16 ----
  const float rsq = rsqrtf(1.0f + EPS_BN);
  auto emit = [&](const v8f& acc, int cobase, int pix) {
    if (cobase + 8 <= Cout) {
      v8bf s;
#pragma unroll
      for (int r = 0; r < 8; ++r) {
        float u = acc[r] + bias[cobase + r];
        if (do_bn) u = u * (gamma[cobase + r] * rsq) + beta[cobase + r];
        if (do_relu) u = fmaxf(u, 0.0f);
        s[r] = (__bf16)u;
      }
      *(v8bf*)(out + (size_t)pix * Cout + cobase) = s;  // 16B aligned (Cout%64==0)
    } else {
#pragma unroll
      for (int r = 0; r < 8; ++r) {
        int co = cobase + r;
        if (co < Cout) {
          float u = acc[r] + bias[co];
          if (do_bn) u = u * (gamma[co] * rsq) + beta[co];
          if (do_relu) u = fmaxf(u, 0.0f);
          out[(size_t)pix * Cout + co] = f2bf(u);
        }
      }
    }
  };
  const int cob0 = co0 + wrow * 32 + g * 8;        // C-layout: VGPR r -> M = r+8g
  const int cob1 = cob0 + 16;
  emit(acc00, cob0, p0);
  emit(acc01, cob0, p1);
  emit(acc10, cob1, p0);
  emit(acc11, cob1, p1);
}

// ---------------------------------------------------------------------------
// Pointwise helpers, NHWC layout (bandwidth-trivial at these sizes)
// ---------------------------------------------------------------------------
__global__ void maxpool2_bf(const bfbits* __restrict__ in, bfbits* __restrict__ out,
                            int C, int H, int W) {
  int Ho = H >> 1, Wo = W >> 1;
  int n = C * Ho * Wo;
  int i = blockIdx.x * blockDim.x + threadIdx.x;
  if (i >= n) return;
  int c  = i % C;
  int sp = i / C;
  int x = sp % Wo, y = sp / Wo;
  const bfbits* p = in + ((size_t)(2 * y) * W + 2 * x) * C + c;
  size_t rs = (size_t)W * C;
  float m = fmaxf(fmaxf(bf2f(p[0]), bf2f(p[C])), fmaxf(bf2f(p[rs]), bf2f(p[rs + C])));
  out[i] = f2bf(m);
}

__global__ void upsample2x_bf(const bfbits* __restrict__ in, bfbits* __restrict__ out,
                              int C, int H, int W) {
  int Ho = H << 1, Wo = W << 1;
  int n = C * Ho * Wo;
  int i = blockIdx.x * blockDim.x + threadIdx.x;
  if (i >= n) return;
  int c  = i % C;
  int sp = i / C;
  int x = sp % Wo, y = sp / Wo;
  out[i] = in[((size_t)(y >> 1) * W + (x >> 1)) * C + c];
}

// map[pix][0:1024]    = fA
// map[pix][1024:2048] = fA * boxsum_4x4(fB)   (rows y-2..y+1, cols x-2..x+1)
__global__ void build_map_bf(const bfbits* __restrict__ fA, const bfbits* __restrict__ fB,
                             bfbits* __restrict__ map, int H, int W) {
  const int C = 1024;
  int n = H * W * 2 * C;
  int i = blockIdx.x * blockDim.x + threadIdx.x;
  if (i >= n) return;
  int c   = i % (2 * C);
  int pix = i / (2 * C);
  if (c < C) {
    map[i] = fA[(size_t)pix * C + c];
  } else {
    int cc = c - C;
    int x = pix % W, y = pix / W;
    float s = 0.0f;
#pragma unroll
    for (int dy = -2; dy <= 1; ++dy) {
      int yy = y + dy;
      if ((unsigned)yy >= (unsigned)H) continue;
#pragma unroll
      for (int dx = -2; dx <= 1; ++dx) {
        int xx = x + dx;
        if ((unsigned)xx >= (unsigned)W) continue;
        s += bf2f(fB[((size_t)yy * W + xx) * C + cc]);
      }
    }
    map[i] = f2bf(bf2f(fA[(size_t)pix * C + cc]) * s);
  }
}

// softmax over C=2 (NHWC bf16 in) -> NCHW f32 out
__global__ void softmax2_bf(const bfbits* __restrict__ in, float* __restrict__ out, int HW) {
  int p = blockIdx.x * blockDim.x + threadIdx.x;
  if (p >= HW) return;
  float a = bf2f(in[2 * p]);
  float b = bf2f(in[2 * p + 1]);
  float m = fmaxf(a, b);
  float ea = __expf(a - m);
  float eb = __expf(b - m);
  float s = ea + eb;
  out[p]      = ea / s;
  out[HW + p] = eb / s;
}

// ---------------------------------------------------------------------------
// Host orchestration
// ---------------------------------------------------------------------------
namespace {

struct LayerDef { int cinSrc, cinP, cout, coutP, pIdx, bn, flip; };

inline void launch_conv(hipStream_t s, const bfbits* in, const bfbits* wt,
                        const bfbits* zbuf, const float* bias, const float* gamma,
                        const float* beta, bfbits* out, int Cin, int Cout, int CoutP,
                        int H, int bn, int relu) {
  int HW = H * H;
  dim3 grid(HW / 128, (Cout + 63) / 64);
  conv3x3_wmma_nhwc<<<grid, 256, 0, s>>>(in, wt, zbuf, bias, gamma, beta, out,
                                         Cin, Cout, CoutP, H, H, bn, relu);
}

} // namespace

extern "C" void kernel_launch(void* const* d_in, const int* in_sizes, int n_in,
                              void* d_out, int out_size, void* d_ws, size_t ws_size,
                              hipStream_t stream) {
  (void)in_sizes; (void)n_in; (void)out_size; (void)ws_size;

  // d_in layout: [0]=imageA [1]=imageB, vgg (w,b)x13 at 2.., enc2 (w,b,g,bt)x2
  // at 28.., dec (w,b,g,bt)x13 at 36..
  static const int vggCin[13]  = {3, 64, 64, 128, 128, 256, 256, 256, 512, 512, 512, 512, 512};
  static const int vggCout[13] = {64, 64, 128, 128, 256, 256, 256, 512, 512, 512, 512, 512, 512};
  static const int enc2Cin[2]  = {512, 1024};
  static const int enc2Cout[2] = {1024, 1024};
  static const int decCin[13]  = {2048, 512, 512, 512, 512, 512, 512, 256, 256, 256, 128, 128, 64};
  static const int decCout[13] = {512, 512, 512, 512, 512, 512, 256, 256, 256, 128, 128, 64, 2};

  LayerDef L[28];
  int nl = 0;
  for (int i = 0; i < 13; ++i) {
    int cp = (vggCin[i] == 3) ? 32 : vggCin[i];          // pad image channels
    L[nl++] = {vggCin[i], cp, vggCout[i], (vggCout[i] + 63) & ~63, 2 + 2 * i, 0, 0};
  }
  for (int j = 0; j < 2; ++j)
    L[nl++] = {enc2Cin[j], enc2Cin[j], enc2Cout[j], (enc2Cout[j] + 63) & ~63, 28 + 4 * j, 1, 0};
  for (int j = 0; j < 13; ++j)
    L[nl++] = {decCin[j], decCin[j], decCout[j], (decCout[j] + 63) & ~63, 36 + 4 * j, 1, 1};

  // ---- workspace bump allocator ------------------------------------------
  char* ws = (char*)d_ws;
  size_t off = 0;
  auto alloc = [&](size_t bytes) -> char* {
    off = (off + 255) & ~(size_t)255;
    char* p = ws + off;
    off += bytes;
    return p;
  };

  // zero page for out-of-image taps (max Cin = 2048 elems)
  bfbits* zbuf = (bfbits*)alloc((size_t)2048 * 2);
  zero_bf<<<(2048 + 255) / 256, 256, 0, stream>>>(zbuf, 2048);

  // bf16 weight table (tap-major, channel-padded on both axes)
  bfbits* wtab[28];
  for (int l = 0; l < 28; ++l) {
    int n = 9 * L[l].coutP * L[l].cinP;
    wtab[l] = (bfbits*)alloc((size_t)n * 2);
    prep_weights<<<(n + 255) / 256, 256, 0, stream>>>(
        (const float*)d_in[L[l].pIdx], wtab[l], L[l].cinSrc, L[l].cinP, L[l].cout,
        L[l].coutP, L[l].flip, n);
  }

  // activation ping-pong (largest: 128 ch @ 512x512 = 33.6M bf16 elems)
  const size_t actElems = (size_t)128 * 512 * 512;
  bfbits* pp[2];
  pp[0] = (bfbits*)alloc(actElems * 2);
  pp[1] = (bfbits*)alloc(actElems * 2);
  bfbits* fA   = (bfbits*)alloc((size_t)1024 * 256 * 2);
  bfbits* fB   = (bfbits*)alloc((size_t)1024 * 256 * 2);
  bfbits* mapA = (bfbits*)alloc((size_t)2048 * 256 * 2);
  bfbits* mapB = (bfbits*)alloc((size_t)2048 * 256 * 2);

  // ---- encoder: VGG16 features + 2-layer BN head -------------------------
  auto run_encoder = [&](const float* img, bfbits* fOut) {
    bfbits* cur = pp[0];
    bfbits* nxt = pp[1];
    int HW0 = 512 * 512;
    img_to_nhwc<<<(HW0 * 32 + 255) / 256, 256, 0, stream>>>(img, cur, HW0);
    int H = 512;
    for (int i = 0; i < 13; ++i) {
      const LayerDef& d = L[i];
      launch_conv(stream, cur, wtab[i], zbuf, (const float*)d_in[d.pIdx + 1],
                  nullptr, nullptr, nxt, d.cinP, d.cout, d.coutP, H, /*bn=*/0, /*relu=*/1);
      { bfbits* t = cur; cur = nxt; nxt = t; }
      if (i == 1 || i == 3 || i == 6 || i == 9 || i == 12) {
        int n = d.cout * (H / 2) * (H / 2);
        maxpool2_bf<<<(n + 255) / 256, 256, 0, stream>>>(cur, nxt, d.cout, H, H);
        H >>= 1;
        bfbits* t = cur; cur = nxt; nxt = t;
      }
    }
    // encoder_l2 (H == 16)
    for (int j = 0; j < 2; ++j) {
      const LayerDef& d = L[13 + j];
      bfbits* dst = (j == 1) ? fOut : nxt;
      launch_conv(stream, cur, wtab[13 + j], zbuf, (const float*)d_in[d.pIdx + 1],
                  (const float*)d_in[d.pIdx + 2], (const float*)d_in[d.pIdx + 3],
                  dst, d.cinP, d.cout, d.coutP, 16, /*bn=*/1, /*relu=*/1);
      if (j == 0) { bfbits* t = cur; cur = nxt; nxt = t; }
    }
  };

  run_encoder((const float*)d_in[0], fA);
  run_encoder((const float*)d_in[1], fB);

  // ---- correlation + concat (NHWC) ---------------------------------------
  {
    int n = 16 * 16 * 2048;
    build_map_bf<<<(n + 255) / 256, 256, 0, stream>>>(fA, fB, mapA, 16, 16);
    build_map_bf<<<(n + 255) / 256, 256, 0, stream>>>(fB, fA, mapB, 16, 16);
  }

  // ---- decoder ------------------------------------------------------------
  auto run_decoder = [&](bfbits* map, float* outp) {
    int t = 0;
    bfbits* cur = map;
    int H = 16;
    int C = 2048;
    for (int j = 0; j < 13; ++j) {
      if (j == 0 || j == 3 || j == 6 || j == 9 || j == 11) {
        int n = C * (2 * H) * (2 * H);
        upsample2x_bf<<<(n + 255) / 256, 256, 0, stream>>>(cur, pp[t], C, H, H);
        H <<= 1;
        cur = pp[t];
        t ^= 1;
      }
      const LayerDef& d = L[15 + j];
      launch_conv(stream, cur, wtab[15 + j], zbuf, (const float*)d_in[d.pIdx + 1],
                  (const float*)d_in[d.pIdx + 2], (const float*)d_in[d.pIdx + 3],
                  pp[t], d.cinP, d.cout, d.coutP, H, /*bn=*/1, /*relu=*/1);
      cur = pp[t];
      t ^= 1;
      C = d.cout;
    }
    int HW = 512 * 512;
    softmax2_bf<<<(HW + 255) / 256, 256, 0, stream>>>(cur, outp, HW);
  };

  float* outA = (float*)d_out;
  float* outB = outA + (size_t)2 * 512 * 512;
  run_decoder(mapA, outA);
  run_decoder(mapB, outB);
}
